// SophDeepseekV3FusedMoE_79224966742832
// MI455X (gfx1250) — matrix operand order repaired and checked by reading, not compiled, not run on hardware
//
#include <hip/hip_runtime.h>
#include <hip/hip_bf16.h>
#include <cstdint>

// Problem constants (DeepSeek-V3-ish fused MoE)
constexpr int NE   = 16;    // experts
constexpr int NT   = 512;   // tokens
constexpr int NH   = 2048;  // hidden
constexpr int NI   = 1408;  // intermediate
constexpr int TOPK = 6;
constexpr int BLK  = 128;   // quant block

// WMMA types (CDNA5 wave32)
typedef __bf16 bf16_t;
typedef bf16_t   v16bf __attribute__((ext_vector_type(16)));
typedef float    v8f   __attribute__((ext_vector_type(8)));
typedef uint32_t u32x8 __attribute__((ext_vector_type(8)));

// f32 -> bf16 bits, round-half-up (one add; the tie-to-even bit costs 2 extra
// VALU per element and is far below bf16 quantization noise here).
__device__ __forceinline__ uint16_t f2bfu(float f) {
    uint32_t u = __builtin_bit_cast(uint32_t, f);
    return (uint16_t)((u + 0x8000u) >> 16);
}

// Pack two f32 into a bf16x2 DWORD: 2x v_add + 1x v_perm_b32
// (selector 0x07060302 takes the high halves of {a,b}; a -> low, b -> high).
__device__ __forceinline__ uint32_t pk_bf16(float a, float b) {
    uint32_t ua = __builtin_bit_cast(uint32_t, a) + 0x8000u;
    uint32_t ub = __builtin_bit_cast(uint32_t, b) + 0x8000u;
    return __builtin_amdgcn_perm(ub, ua, 0x07060302u);
}

// A-matrix fragment (16x32 bf16) straight from two 16B LDS beats; the v16bf is
// bit-identical to 8 DWORDs, so a bit_cast avoids any per-element v_mov_b16.
// Contiguous K layout: lanes 0-15 hold K = k0+{0..15}; lanes 16-31 k0+{16..31}.
__device__ __forceinline__ v16bf load_a_lds(const uint16_t* row, int k0, int g) {
    const uint4 u0 = *(const uint4*)(row + k0 + g * 16);
    const uint4 u1 = *(const uint4*)(row + k0 + g * 16 + 8);
    u32x8 u = {u0.x, u0.y, u0.z, u0.w, u1.x, u1.y, u1.z, u1.w};
    return __builtin_bit_cast(v16bf, u);
}

// B-matrix fragment (32x16 bf16) from a row-major weight row (N-row, K contig):
// lane holds column N = lane&15; lanes 0-15 K = kk..kk+15, lanes 16-31 kk+16..31
// (kk folded by caller). 16 consecutive f32 -> 4x global_load_b128; the 128x128
// block inverse-scale is folded during the f32->bf16 convert (fused dequant).
__device__ __forceinline__ v16bf load_b_dequant(const float* wrow, float s) {
    const float4* p = (const float4*)wrow;
    float4 f0 = p[0], f1 = p[1], f2 = p[2], f3 = p[3];
    u32x8 u;
    u[0] = pk_bf16(f0.x * s, f0.y * s);
    u[1] = pk_bf16(f0.z * s, f0.w * s);
    u[2] = pk_bf16(f1.x * s, f1.y * s);
    u[3] = pk_bf16(f1.z * s, f1.w * s);
    u[4] = pk_bf16(f2.x * s, f2.y * s);
    u[5] = pk_bf16(f2.z * s, f2.w * s);
    u[6] = pk_bf16(f3.x * s, f3.y * s);
    u[7] = pk_bf16(f3.z * s, f3.w * s);
    return __builtin_bit_cast(v16bf, u);
}

// ---------------------------------------------------------------------------
// Kernel 1: build per-expert token lists from top-k routing.
// ---------------------------------------------------------------------------
__global__ void route_kernel(const int* __restrict__ sel,
                             const float* __restrict__ rw,
                             int* __restrict__ counts,
                             int* __restrict__ tok_idx,
                             float* __restrict__ tok_w) {
    __shared__ int lcnt[NE];
    const int tid = threadIdx.x;
    if (tid < NE) lcnt[tid] = 0;
    __syncthreads();
    for (int i = tid; i < NT * TOPK; i += blockDim.x) {
        const int e = sel[i];
        const int pos = atomicAdd(&lcnt[e], 1);
        tok_idx[e * NT + pos] = i / TOPK;
        tok_w[e * NT + pos]   = rw[i];
    }
    __syncthreads();
    if (tid < NE) counts[tid] = lcnt[tid];
}

// ---------------------------------------------------------------------------
// Kernel 2: zero the output (experts scatter-add into it).
// ---------------------------------------------------------------------------
__global__ void zero_out(float* __restrict__ o, int n4) {
    const int i = blockIdx.x * blockDim.x + threadIdx.x;
    if (i < n4) ((float4*)o)[i] = make_float4(0.f, 0.f, 0.f, 0.f);
}

// ---------------------------------------------------------------------------
// Kernel 3: fused dequant + gate/up GEMM + SwiGLU + down GEMM + combine.
// Block = 256 threads (8 wave32s), grid = (M-tiles of 16 tokens, experts).
// LDS: x tile (16x2048 bf16, 65KB) + h tile (16x1408 bf16, 45KB) — uses the
// CDNA5 320KB-per-WGP LDS budget to keep the whole SwiGLU intermediate on-chip.
// ---------------------------------------------------------------------------
constexpr int XS_STRIDE = NH + 32;   // 2080, 16B-aligned rows, bank-skewed
constexpr int HS_STRIDE = NI + 32;   // 1440

__global__ __launch_bounds__(256)
void moe_kernel(const float* __restrict__ x,
                const float* __restrict__ gw, const float* __restrict__ gsc,
                const float* __restrict__ uw, const float* __restrict__ usc,
                const float* __restrict__ dw, const float* __restrict__ dsc,
                const int* __restrict__ counts,
                const int* __restrict__ tok_idx,
                const float* __restrict__ tok_w,
                float* __restrict__ out) {
    __shared__ uint16_t xs[16 * XS_STRIDE];   // x tile, bf16 bits
    __shared__ uint16_t hs[16 * HS_STRIDE];   // silu(g)*u tile, bf16 bits
    __shared__ int   s_tok[16];
    __shared__ float s_rw[16];

    const int e   = blockIdx.y;
    const int m16 = blockIdx.x * 16;
    const int cnt = counts[e];
    if (m16 >= cnt) return;                   // uniform per block: EXEC stays full

    const int tid  = threadIdx.x;
    const int lane = tid & 31;
    const int wv   = tid >> 5;
    const int g    = lane >> 4;               // lane group (K-half select)
    const int nloc = lane & 15;               // output column within 16-wide tile

    // --- gather token list (pad with token 0 / weight 0) ---
    if (tid < 16) {
        const int idx = m16 + tid;
        const bool ok = idx < cnt;
        s_tok[tid] = ok ? tok_idx[e * NT + idx] : 0;
        s_rw[tid]  = ok ? tok_w[e * NT + idx] : 0.f;
    }
    __syncthreads();

    // --- stage x rows for the 16 tokens into LDS as bf16 ---
    for (int i = tid; i < 16 * (NH / 4); i += 256) {
        const int r = i >> 9;                 // NH/4 == 512 float4 per row
        const int c = i & 511;
        const float4 v = ((const float4*)(x + (size_t)s_tok[r] * NH))[c];
        *(uint2*)(xs + r * XS_STRIDE + c * 4) =
            make_uint2(pk_bf16(v.x, v.y), pk_bf16(v.z, v.w));
    }
    __syncthreads();

    // =========== Phase 1: h = silu(x*Wg^T) * (x*Wu^T), kept in LDS ===========
    const uint16_t* arow = xs + (size_t)(lane & 15) * XS_STRIDE;  // A row M=lane&15
    for (int ct = wv; ct < NI / 16; ct += 8) {
        const int n = ct * 16 + nloc;
        const float* gw_row = gw + ((size_t)e * NI + n) * NH;
        const float* uw_row = uw + ((size_t)e * NI + n) * NH;
        const float* gs_row = gsc + ((size_t)e * (NI / BLK) + (n >> 7)) * (NH / BLK);
        const float* us_row = usc + ((size_t)e * (NI / BLK) + (n >> 7)) * (NH / BLK);
        v8f accg = {}, accu = {};
        #pragma unroll 2
        for (int k0 = 0; k0 < NH; k0 += 32) {
            const int kk = k0 + g * 16;
            __builtin_prefetch(gw_row + kk + 512, 0, 0);   // stream-ahead hint
            const v16bf a  = load_a_lds(arow, k0, g);
            const v16bf bg = load_b_dequant(gw_row + kk, gs_row[kk >> 7]);
            const v16bf bu = load_b_dequant(uw_row + kk, us_row[kk >> 7]);
            accg = __builtin_amdgcn_wmma_f32_16x16x32_bf16(
                       false, a, false, bg, (short)0, accg, false, false);
            accu = __builtin_amdgcn_wmma_f32_16x16x32_bf16(
                       false, a, false, bu, (short)0, accu, false, false);
        }
        // SwiGLU with hardware reciprocal: silu(g) = g * v_rcp(1 + exp(-g)).
        // C/D layout: VGPR r, lane -> M = r + 8*g, N = nloc.
        #pragma unroll
        for (int r = 0; r < 8; ++r) {
            const int m = r + g * 8;
            const float gv  = accg[r];
            const float sig = __builtin_amdgcn_rcpf(1.f + __expf(-gv));
            const float hv  = gv * sig * accu[r];
            hs[m * HS_STRIDE + ct * 16 + nloc] = f2bfu(hv);
        }
    }
    __syncthreads();

    // =========== Phase 2: out[t] += rw * (h * Wd^T) ===========
    const uint16_t* hrow = hs + (size_t)(lane & 15) * HS_STRIDE;
    for (int nt = wv; nt < NH / 16; nt += 8) {
        const int n = nt * 16 + nloc;
        const float* dw_row = dw + ((size_t)e * NH + n) * NI;
        const float* ds_row = dsc + ((size_t)e * (NH / BLK) + (n >> 7)) * (NI / BLK);
        v8f acc = {};
        #pragma unroll 2
        for (int k0 = 0; k0 < NI; k0 += 32) {
            const int kk = k0 + g * 16;
            __builtin_prefetch(dw_row + kk + 512, 0, 0);
            const v16bf a = load_a_lds(hrow, k0, g);
            const v16bf b = load_b_dequant(dw_row + kk, ds_row[kk >> 7]);
            acc = __builtin_amdgcn_wmma_f32_16x16x32_bf16(
                      false, a, false, b, (short)0, acc, false, false);
        }
        #pragma unroll
        for (int r = 0; r < 8; ++r) {
            const int m = r + g * 8;
            atomicAdd(&out[(size_t)s_tok[m] * NH + n], acc[r] * s_rw[m]);
        }
    }
}

// ---------------------------------------------------------------------------
extern "C" void kernel_launch(void* const* d_in, const int* in_sizes, int n_in,
                              void* d_out, int out_size, void* d_ws, size_t ws_size,
                              hipStream_t stream) {
    (void)in_sizes; (void)n_in; (void)out_size; (void)ws_size;
    const float* x   = (const float*)d_in[0];
    const float* gw  = (const float*)d_in[1];
    const float* gsc = (const float*)d_in[2];
    const float* uw  = (const float*)d_in[3];
    const float* usc = (const float*)d_in[4];
    const float* dw  = (const float*)d_in[5];
    const float* dsc = (const float*)d_in[6];
    const float* rw  = (const float*)d_in[7];
    const int*   sel = (const int*)d_in[8];
    float* out = (float*)d_out;

    // Workspace layout: counts[E] | tok_idx[E*T] | tok_w[E*T]  (~64.1 KB)
    int*   counts  = (int*)d_ws;
    int*   tok_idx = counts + NE;
    float* tok_w   = (float*)(tok_idx + NE * NT);

    zero_out<<<dim3((NT * NH / 4 + 255) / 256), dim3(256), 0, stream>>>(out, NT * NH / 4);
    route_kernel<<<dim3(1), dim3(256), 0, stream>>>(sel, rw, counts, tok_idx, tok_w);
    moe_kernel<<<dim3(NT / 16, NE), dim3(256), 0, stream>>>(
        x, gw, gsc, uw, usc, dw, dsc, counts, tok_idx, tok_w, out);
}
